// ReverseRNNLayer_76089640616411
// MI455X (gfx1250) — compile-verified, matrix operand order checked by reading, not run on hardware
//
#include <hip/hip_runtime.h>
#include <hip/hip_bf16.h>
#include <math.h>

// ReverseRNNLayer for MI455X (gfx1250, wave32, WMMA bf16 16x16x32, f32 accum)
//
// Phase 0: convert Wih/Whh to bf16 in d_ws (256 KB).
// Phase 1: xp = x . Wih^T + bih  (131072 x 256 x 256 GEMM), rows follow the
//          per-chunk reversed processing order; results are written directly
//          into d_out at their FINAL output location s_out = (C-1-c)*L+(L-1-l).
// Phase 2: 256-step recurrence h = tanh(xp_t + h . Whh^T + bhh) over 512
//          independent rows; each step reads xp from d_out and overwrites the
//          same slot with h (in-place), h kept bf16 in double-buffered LDS,
//          Whh^T fragments hoisted into registers (time-invariant).

typedef __bf16 bf16;
typedef __attribute__((ext_vector_type(16))) __bf16 v16bf;
typedef __attribute__((ext_vector_type(8)))  __bf16 v8bf;
typedef __attribute__((ext_vector_type(4)))  __bf16 v4bf;
typedef __attribute__((ext_vector_type(8)))  float  v8f;

#define Bsz 64
#define Ssz 2048
#define Fsz 256
#define Hsz 256
#define Lsz 256
#define Csz 8
#define LDH 264   // padded LDS row stride in halfs: 528B = 132 DW, 4 DW/bank step
                  // -> 16 lanes of a b128 fragment load cover all 64 banks

// --- 16x32 bf16 fragment (A) / 32x16 bf16 fragment (B) loader -----------------
// caller passes p = base + row*stride + kstep*32 + 8*(lane>>4);
// halfs 0..7 are contiguous at p, halfs 8..15 contiguous at p+16 (ISA 7.12.2).
static __device__ __forceinline__ v16bf frag16(const bf16* p) {
  v8bf lo = *(const v8bf*)(p);
  v8bf hi = *(const v8bf*)(p + 16);
  return __builtin_shufflevector(lo, hi, 0,1,2,3,4,5,6,7,8,9,10,11,12,13,14,15);
}

static __device__ __forceinline__ v8f wmma_bf16(v16bf a, v16bf b, v8f c) {
  return __builtin_amdgcn_wmma_f32_16x16x32_bf16(false, a, false, b,
                                                 (short)0, c, false, false);
}

static __device__ __forceinline__ float fast_tanh(float x) {
#if __has_builtin(__builtin_amdgcn_tanhf)
  return __builtin_amdgcn_tanhf(x);       // v_tanh_f32 (gfx1250 TRANS op)
#else
  return ::tanhf(x);
#endif
}

// ============================ Phase 0: weight convert ========================
__global__ void __launch_bounds__(256) k_cvt_weights(const float* __restrict__ wih,
                                                     const float* __restrict__ whh,
                                                     bf16* __restrict__ wihb,
                                                     bf16* __restrict__ whhb) {
  int i = blockIdx.x * 256 + threadIdx.x;   // 65536 elements each
  wihb[i] = (bf16)wih[i];
  whhb[i] = (bf16)whh[i];
}

// ============================ Phase 1: xp GEMM ===============================
// grid = 131072/64 WGs, 256 threads (8 waves). WG tile: 64 rows x 256 cols.
__global__ void __launch_bounds__(256) k_xp_gemm(const float* __restrict__ x,
                                                 const bf16* __restrict__ wihb,
                                                 const float* __restrict__ bih,
                                                 float* __restrict__ out) {
  extern __shared__ bf16 sA[];              // 64 x LDH halfs (33792 B)
  const int tid = threadIdx.x;
  const int wg  = blockIdx.x;

  // ---- stage A tile: 64 rows of x (f32 -> bf16), rows in processing order
#pragma unroll
  for (int it = 0; it < 16; ++it) {
    int i = it * 256 + tid;                 // 0..4095 (row, float4-chunk)
    int r = i >> 6;
    int q = i & 63;
    int m = wg * 64 + r;                    // m = (b*C + c)*L + l
    int b = m >> 11, rem = m & 2047, c = rem >> 8, l = rem & 255;
    int sin = c * Lsz + (Lsz - 1 - l);      // reversed time within chunk
    const float4 v = *(const float4*)(x + ((size_t)(b * Ssz + sin) << 8) + (q << 2));
    v4bf bv;
    bv.x = (bf16)v.x; bv.y = (bf16)v.y; bv.z = (bf16)v.z; bv.w = (bf16)v.w;
    *(v4bf*)(sA + r * LDH + (q << 2)) = bv;
  }
  __syncthreads();

  const int lane  = tid & 31;
  const int wave  = tid >> 5;
  const int l15   = lane & 15;
  const int kfold = (lane >> 4) << 3;       // 0 or 8
  const int ncb   = wave << 5;              // this wave's 32-column base

  v8f acc[4][2];
#pragma unroll
  for (int mt = 0; mt < 4; ++mt) { acc[mt][0] = {}; acc[mt][1] = {}; }

#pragma unroll
  for (int ks = 0; ks < 8; ++ks) {
    // B(K,n) = Wih[n][K]  (bf16, row-major, L2-resident)
    const bf16* bp = wihb + (size_t)(ncb + l15) * Fsz + ks * 32 + kfold;
    v16bf b0 = frag16(bp);
    v16bf b1 = frag16(bp + 16 * Fsz);
#pragma unroll
    for (int mt = 0; mt < 4; ++mt) {
      v16bf a = frag16(sA + (mt * 16 + l15) * LDH + ks * 32 + kfold);
      acc[mt][0] = wmma_bf16(a, b0, acc[mt][0]);
      acc[mt][1] = wmma_bf16(a, b1, acc[mt][1]);
    }
  }

  const float bi0 = bih[ncb + l15];
  const float bi1 = bih[ncb + 16 + l15];
#pragma unroll
  for (int mt = 0; mt < 4; ++mt)
#pragma unroll
    for (int v = 0; v < 8; ++v) {
      int rl = mt * 16 + v + kfold;         // D-matrix row mapping
      int m  = wg * 64 + rl;
      int b = m >> 11, rem = m & 2047, c = rem >> 8, l = rem & 255;
      size_t ro = (size_t)(b * Ssz + (Csz - 1 - c) * Lsz + (Lsz - 1 - l)) << 8;
      out[ro + ncb + l15]      = acc[mt][0][v] + bi0;
      out[ro + ncb + 16 + l15] = acc[mt][1][v] + bi1;
    }
}

// ============================ Phase 2: recurrence ============================
// grid = 512/16 = 32 WGs, 256 threads (8 waves). WG owns 16 (b,c) rows.
__global__ void __launch_bounds__(256) k_rnn_scan(float* __restrict__ out,
                                                  const bf16* __restrict__ whhb,
                                                  const float* __restrict__ bhh) {
  extern __shared__ bf16 sH[];              // 2 x 16 x LDH halfs (16896 B)
  const int tid = threadIdx.x;
  const int g   = blockIdx.x;

  for (int i = tid; i < 2 * 16 * LDH; i += 256) sH[i] = (bf16)0.0f;  // h0 = 0
  __syncthreads();

  const int lane  = tid & 31;
  const int wave  = tid >> 5;
  const int l15   = lane & 15;
  const int kfold = (lane >> 4) << 3;
  const int ncb   = wave << 5;
  const int col0  = ncb + l15, col1 = ncb + 16 + l15;

  // Hoist Whh^T fragments: B(K,n) = Whh[n][K]; invariant over all 256 steps.
  v16bf bfr[2][8];
#pragma unroll
  for (int nt = 0; nt < 2; ++nt)
#pragma unroll
    for (int ks = 0; ks < 8; ++ks)
      bfr[nt][ks] = frag16(whhb + (size_t)(ncb + nt * 16 + l15) * Hsz
                                 + ks * 32 + kfold);

  const float bh0 = bhh[col0];
  const float bh1 = bhh[col1];

  // Per-row output base pointers (row p = b*C + c; rows 0-7 -> b=2g, 8-15 -> 2g+1)
  float* prow[8];
  int hidok[8], bidx[8];
#pragma unroll
  for (int v = 0; v < 8; ++v) {
    int rl = v + kfold;
    int p  = g * 16 + rl;
    int b  = p >> 3, c = p & 7;
    prow[v]  = out + ((size_t)(b * Ssz + (Csz - 1 - c) * Lsz + (Lsz - 1)) << 8);
    hidok[v] = (c == Csz - 1);
    bidx[v]  = b;
  }
  float* hid = out + (size_t)Bsz * Ssz * Hsz;   // hidden_final (B x H) tail

  for (int l = 0; l < Lsz; ++l) {
    const bf16* hr = sH + (l & 1) * (16 * LDH);
    bf16*       hw = sH + ((l & 1) ^ 1) * (16 * LDH);

    // accumulator init = xp (read in place from d_out, step-l slot)
    v8f a0 = {}, a1 = {};
#pragma unroll
    for (int v = 0; v < 8; ++v) {
      const float* pr = prow[v] - ((size_t)l << 8);
      a0[v] = pr[col0];
      a1[v] = pr[col1];
    }

#pragma unroll
    for (int ks = 0; ks < 8; ++ks) {
      v16bf a = frag16(hr + l15 * LDH + ks * 32 + kfold);
      a0 = wmma_bf16(a, bfr[0][ks], a0);
      a1 = wmma_bf16(a, bfr[1][ks], a1);
    }

#pragma unroll
    for (int v = 0; v < 8; ++v) {
      float t0 = fast_tanh(a0[v] + bh0);
      float t1 = fast_tanh(a1[v] + bh1);
      float* pr = prow[v] - ((size_t)l << 8);
      pr[col0] = t0;                          // overwrite xp slot with h (f32)
      pr[col1] = t1;
      int rl = v + kfold;
      hw[rl * LDH + col0] = (bf16)t0;         // h for next step (bf16 LDS)
      hw[rl * LDH + col1] = (bf16)t1;
      if ((l == Lsz - 1) && hidok[v]) {       // tail chunk's final hidden state
        hid[bidx[v] * Hsz + col0] = t0;
        hid[bidx[v] * Hsz + col1] = t1;
      }
    }
    __syncthreads();                          // h visible before next step reads
  }
}

// ================================ launcher ===================================
extern "C" void kernel_launch(void* const* d_in, const int* in_sizes, int n_in,
                              void* d_out, int out_size, void* d_ws, size_t ws_size,
                              hipStream_t stream) {
  (void)in_sizes; (void)n_in; (void)out_size; (void)ws_size;
  const float* x   = (const float*)d_in[0];
  // d_in[1] = hidden_state: unused (reference resets h to zero per chunk)
  const float* wih = (const float*)d_in[2];
  const float* whh = (const float*)d_in[3];
  const float* bih = (const float*)d_in[4];
  const float* bhh = (const float*)d_in[5];
  float* out = (float*)d_out;

  bf16* wihb = (bf16*)d_ws;                 // 128 KB
  bf16* whhb = wihb + Hsz * Fsz;            // 128 KB

  k_cvt_weights<<<(Hsz * Fsz) / 256, 256, 0, stream>>>(wih, whh, wihb, whhb);
  k_xp_gemm<<<(Bsz * Csz * Lsz) / 64, 256,
              64 * LDH * (int)sizeof(bf16), stream>>>(x, wihb, bih, out);
  k_rnn_scan<<<(Bsz * Csz) / 16, 256,
               2 * 16 * LDH * (int)sizeof(bf16), stream>>>(out, whhb, bhh);
}
